// MultiheadAttention_11029476016138
// MI455X (gfx1250) — matrix operand order repaired and checked by reading, not compiled, not run on hardware
//
#include <hip/hip_runtime.h>
#include <hip/hip_bf16.h>

// Flash-attention forward for B=2, H=16, S=2048, hd=64, fp32 in/out.
// S^T = K·Q^T and O^T = V^T·P^T via v_wmma_f32_16x16x32_f16 (wave32), so the
// softmax statistics are per-lane scalars (query index == lane%16 in the WMMA
// C/D layout). Double-buffered LDS, software-pipelined staging, ds_swizzle
// SWAPX16 for all half-wave exchanges.

typedef _Float16 v16h __attribute__((ext_vector_type(16)));
typedef _Float16 h8   __attribute__((ext_vector_type(8)));
typedef float    v8f  __attribute__((ext_vector_type(8)));

#define WMMA_F16(A, B, C) \
  __builtin_amdgcn_wmma_f32_16x16x32_f16(false, (A), false, (B), (short)0, (C), false, false)

constexpr int B_  = 2;
constexpr int S_  = 2048;
constexpr int D_  = 1024;
constexpr int H_  = 16;
constexpr int HD  = 64;     // head dim
constexpr int QT  = 64;     // q rows per workgroup (16 per wave, 4 waves)
constexpr int KT  = 32;     // keys per iteration (one 16x32 WMMA K-step)
constexpr int KP  = 72;     // LDS pitch (halves) for K rows   (16B-aligned, bank-spread)
constexpr int VP  = 40;     // LDS pitch (halves) for V^T rows (16B-aligned, bank-spread)
constexpr int NBLK = S_ / KT;
constexpr float SCALE = 0.125f;  // 1/sqrt(64), power of two -> exact in f16

// Half-wave swap (lane L <-> lane L^16): ds_swizzle group-of-32,
// offset = xor_mask<<10 | or_mask<<5 | and_mask = 0x10<<10 | 0x1f = 0x401F (SWAPX16).
__device__ inline float swap16f(float x) {
  return __int_as_float(__builtin_amdgcn_ds_swizzle(__float_as_int(x), 0x401F));
}

// One 16x32 f16 A-fragment row for this lane from LDS.
// A layout: lane L holds M=L%16; element e -> K=2*off8+e.
__device__ inline v16h lds_a_frag(const _Float16* rowbase, int off8) {
  h8 g0 = *(const h8*)(rowbase + 2 * off8);
  h8 g1 = *(const h8*)(rowbase + 2 * off8 + 8);
  return __builtin_shufflevector(g0, g1, 0, 1, 2, 3, 4, 5, 6, 7,
                                 8, 9, 10, 11, 12, 13, 14, 15);
}

// Per-thread staging registers: 16 K halves + 16 V halves (one (key, 16-col chunk)).
struct Stage { h8 k0, k1, v0, v1; };

__device__ inline void stage_load(const float* kr, const float* vr, Stage& st) {
#pragma unroll
  for (int e = 0; e < 8; e += 4) {
    float4 fk = *(const float4*)(kr + e);
    float4 fv = *(const float4*)(vr + e);
    st.k0[e + 0] = (_Float16)fk.x; st.k0[e + 1] = (_Float16)fk.y;
    st.k0[e + 2] = (_Float16)fk.z; st.k0[e + 3] = (_Float16)fk.w;
    st.v0[e + 0] = (_Float16)fv.x; st.v0[e + 1] = (_Float16)fv.y;
    st.v0[e + 2] = (_Float16)fv.z; st.v0[e + 3] = (_Float16)fv.w;
  }
#pragma unroll
  for (int e = 0; e < 8; e += 4) {
    float4 fk = *(const float4*)(kr + 8 + e);
    float4 fv = *(const float4*)(vr + 8 + e);
    st.k1[e + 0] = (_Float16)fk.x; st.k1[e + 1] = (_Float16)fk.y;
    st.k1[e + 2] = (_Float16)fk.z; st.k1[e + 3] = (_Float16)fk.w;
    st.v1[e + 0] = (_Float16)fv.x; st.v1[e + 1] = (_Float16)fv.y;
    st.v1[e + 2] = (_Float16)fv.z; st.v1[e + 3] = (_Float16)fv.w;
  }
}

__device__ inline void stage_store(_Float16* sKb, _Float16* sVb,
                                   int lkey, int lc, const Stage& st) {
  _Float16* kd = &sKb[lkey * KP + lc * 16];
  *(h8*)(kd)     = st.k0;                       // contiguous b128 stores
  *(h8*)(kd + 8) = st.k1;
#pragma unroll
  for (int e = 0; e < 8; ++e)                    // transposed scatter (V^T)
    sVb[(lc * 16 + e) * VP + lkey] = st.v0[e];
#pragma unroll
  for (int e = 0; e < 8; ++e)
    sVb[(lc * 16 + 8 + e) * VP + lkey] = st.v1[e];
}

__global__ __launch_bounds__(128, 1)
void fa_fwd_f16wmma(const float* __restrict__ q,
                    const float* __restrict__ k,
                    const float* __restrict__ v,
                    float* __restrict__ out) {
  // Double-buffered tiles: K [key][hd] f16 (32x64, pitch 72), V^T [hd][key] (64x32, pitch 40).
  __shared__ __align__(16) _Float16 sK[2][KT * KP];
  __shared__ __align__(16) _Float16 sV[2][HD * VP];

  const int tid   = threadIdx.x;
  const int lane  = tid & 31;
  const int wave  = tid >> 5;
  const int ln    = lane & 15;
  const bool lo   = lane < 16;
  const int off8  = lo ? 0 : 8;
  const int off16 = lo ? 0 : 16;

  const int h  = blockIdx.y;
  const int bi = blockIdx.z;
  const int q0 = blockIdx.x * QT;

  // ---- Q^T B-fragments, resident in VGPRs for the whole kernel ----------
  // B layout: lane L holds N=q=L%16; element e -> K(hd) = c*32 + off16 + e.
  const int    qrow = q0 + wave * 16 + ln;
  const float* qptr = q + ((size_t)(bi * S_ + qrow)) * (size_t)D_ + h * HD;
  v16h bQ[2];
#pragma unroll
  for (int c = 0; c < 2; ++c) {
    const float* p = qptr + c * 32 + off16;
#pragma unroll
    for (int e = 0; e < 16; e += 4) {
      float4 f = *(const float4*)(p + e);
      bQ[c][e + 0] = (_Float16)(f.x * SCALE);
      bQ[c][e + 1] = (_Float16)(f.y * SCALE);
      bQ[c][e + 2] = (_Float16)(f.z * SCALE);
      bQ[c][e + 3] = (_Float16)(f.w * SCALE);
    }
  }

  // ---- online-softmax state (per-lane scalars: this lane's q column) ----
  float m = -__builtin_inff();
  float l = 0.0f;
  v8f oacc[4];  // O^T tiles: td covers d = 16*td .. 16*td+15
#pragma unroll
  for (int td = 0; td < 4; ++td)
#pragma unroll
    for (int r = 0; r < 8; ++r) oacc[td][r] = 0.0f;

  // Cooperative staging: one (key, 16-col chunk) per thread.
  const int lkey = tid >> 2;  // 0..31
  const int lc   = tid & 3;   // 0..3
  const float* kptr = k + ((size_t)(bi * S_ + lkey)) * (size_t)D_ + h * HD + lc * 16;
  const float* vptr = v + ((size_t)(bi * S_ + lkey)) * (size_t)D_ + h * HD + lc * 16;

  // Prologue: stage block 0 into buffer 0.
  Stage st;
  stage_load(kptr, vptr, st);
  stage_store(sK[0], sV[0], lkey, lc, st);

  for (int it = 0; it < NBLK; ++it) {
    __syncthreads();  // staging of buf[it&1] complete; buf[(it+1)&1] free to write

    const _Float16* sKb = sK[it & 1];
    const _Float16* sVb = sV[it & 1];
    _Float16* sKn = sK[(it + 1) & 1];
    _Float16* sVn = sV[(it + 1) & 1];

    // Issue next block's global loads early (latency hidden under WMMA+softmax).
    const bool more = (it + 1) < NBLK;
    if (more) {
      const size_t adv = (size_t)(it + 1) * KT * D_;
      stage_load(kptr + adv, vptr + adv, st);
      if (it + 2 < NBLK) {  // pull block it+2 toward the caches (global_prefetch_b8)
        __builtin_prefetch(kptr + adv + (size_t)KT * D_, 0, 3);
        __builtin_prefetch(vptr + adv + (size_t)KT * D_, 0, 3);
      }
    }

    // ---- preload ALL fragments so DS waits batch and WMMAs run back-to-back
    v16h aK0 = lds_a_frag(&sKb[(0  + ln) * KP + 0 ], off8);
    v16h aK1 = lds_a_frag(&sKb[(0  + ln) * KP + 32], off8);
    v16h aK2 = lds_a_frag(&sKb[(16 + ln) * KP + 0 ], off8);
    v16h aK3 = lds_a_frag(&sKb[(16 + ln) * KP + 32], off8);
    v16h aV0 = lds_a_frag(&sVb[(0  + ln) * VP], off8);
    v16h aV1 = lds_a_frag(&sVb[(16 + ln) * VP], off8);
    v16h aV2 = lds_a_frag(&sVb[(32 + ln) * VP], off8);
    v16h aV3 = lds_a_frag(&sVb[(48 + ln) * VP], off8);

    // ---- S^T tiles: (keys 0-15 | 16-31) x (16 q), K-dim = hd = 64 -------
    v8f z;
#pragma unroll
    for (int r = 0; r < 8; ++r) z[r] = 0.0f;
    v8f s0 = WMMA_F16(aK0, bQ[0], z);
    v8f s1 = WMMA_F16(aK2, bQ[0], z);
    s0 = WMMA_F16(aK1, bQ[1], s0);
    s1 = WMMA_F16(aK3, bQ[1], s1);

    // ---- online softmax (reduce over r, + one SWAPX16) ------------------
    float mloc = -__builtin_inff();
#pragma unroll
    for (int r = 0; r < 8; ++r) mloc = fmaxf(mloc, fmaxf(s0[r], s1[r]));
    mloc = fmaxf(mloc, swap16f(mloc));
    const float mnew  = fmaxf(m, mloc);
    const float alpha = __expf(m - mnew);   // first iter: exp(-inf) = 0

    float p0[8], p1[8];
    float ls = 0.0f;
#pragma unroll
    for (int r = 0; r < 8; ++r) {
      p0[r] = __expf(s0[r] - mnew);
      p1[r] = __expf(s1[r] - mnew);
      ls += p0[r] + p1[r];
    }
    ls += swap16f(ls);
    l = l * alpha + ls;
    m = mnew;

    // ---- build P^T B-fragment (keys 0..31 x 16 q) via SWAPX16 -----------
    // B element e: key = (lo?0:16)+e. Source tiles hold keys r+(lo?0:8).
    v16h bP;
#pragma unroll
    for (int r = 0; r < 8; ++r) {
      float x0 = swap16f(p0[r]);
      float x1 = swap16f(p1[r]);
      bP[r]     = (_Float16)(lo ? p0[r] : x1);
      bP[r + 8] = (_Float16)(lo ? x0 : p1[r]);
    }

    // ---- O^T += V^T · P^T  (rescale running O by alpha first) -----------
    v8f c0 = oacc[0], c1 = oacc[1], c2 = oacc[2], c3 = oacc[3];
#pragma unroll
    for (int r = 0; r < 8; ++r) {
      c0[r] *= alpha; c1[r] *= alpha; c2[r] *= alpha; c3[r] *= alpha;
    }
    oacc[0] = WMMA_F16(aV0, bP, c0);
    oacc[1] = WMMA_F16(aV1, bP, c1);
    oacc[2] = WMMA_F16(aV2, bP, c2);
    oacc[3] = WMMA_F16(aV3, bP, c3);

    // ---- store next block's staged tile into the other buffer -----------
    if (more) stage_store(sKn, sVn, lkey, lc, st);
  }

  // ---- epilogue: out[b][q][h*64 + d] = O^T[d][q] / l ---------------------
  const float inv = 1.0f / l;
  float* obase = out + ((size_t)(bi * S_ + qrow)) * (size_t)D_ + h * HD;
#pragma unroll
  for (int td = 0; td < 4; ++td) {
    float4 a, b;
    a.x = oacc[td][0] * inv; a.y = oacc[td][1] * inv;
    a.z = oacc[td][2] * inv; a.w = oacc[td][3] * inv;
    b.x = oacc[td][4] * inv; b.y = oacc[td][5] * inv;
    b.z = oacc[td][6] * inv; b.w = oacc[td][7] * inv;
    float* p = obase + td * 16 + off8;   // d = 16*td + r + (lo?0:8): 8 consecutive floats
    ((float4*)p)[0] = a;
    ((float4*)p)[1] = b;
  }
}

extern "C" void kernel_launch(void* const* d_in, const int* in_sizes, int n_in,
                              void* d_out, int out_size, void* d_ws, size_t ws_size,
                              hipStream_t stream) {
  (void)in_sizes; (void)n_in; (void)out_size; (void)d_ws; (void)ws_size;
  const float* q = (const float*)d_in[0];
  const float* k = (const float*)d_in[1];
  const float* v = (const float*)d_in[2];
  float* out = (float*)d_out;
  dim3 grid(S_ / QT, H_, B_);   // (32, 16, 2) workgroups
  dim3 block(128);              // 4 wave32 waves
  fa_fwd_f16wmma<<<grid, block, 0, stream>>>(q, k, v, out);
}
